// QwenMoEDecoderLayer_62775241998545
// MI455X (gfx1250) — matrix operand order repaired and checked by reading, not compiled, not run on hardware
//
#include <hip/hip_runtime.h>
#include <hip/hip_bf16.h>

// ---------------- problem constants (match reference) ----------------
namespace cfg {
constexpr int S    = 1024;
constexpr int H    = 1024;
constexpr int NH   = 16;
constexpr int KVH  = 4;
constexpr int HD   = 64;
constexpr int REP  = NH / KVH;       // 4
constexpr int E    = 64;
constexpr int TOPK = 8;
constexpr int I    = 1408;
constexpr int IS   = 1408;
constexpr int T    = S;              // B==1
constexpr int CAP  = 2 * (T * TOPK / E);  // 256
}
using namespace cfg;

// ---------------- WMMA types / helpers ----------------
typedef __attribute__((ext_vector_type(16))) __bf16 v16bf;
typedef __attribute__((ext_vector_type(2)))  __bf16 v2bf;
typedef __attribute__((ext_vector_type(8)))  float  v8f;

__device__ __forceinline__ unsigned short f2bfu(float f) {
  return __builtin_bit_cast(unsigned short, (__bf16)f);   // native cvt, RNE
}
__device__ __forceinline__ unsigned pack2bf(float a, float b) {
  v2bf v = { (__bf16)a, (__bf16)b };                      // -> v_cvt_pk_bf16_f32
  return __builtin_bit_cast(unsigned, v);
}
__device__ __forceinline__ float silu(float x) {
  return x * (1.0f / (1.0f + __expf(-x)));
}

__device__ __forceinline__ v8f wmma_bf16(v16bf a, v16bf b, v8f c) {
  return __builtin_amdgcn_wmma_f32_16x16x32_bf16(
      /*neg_a=*/false, a, /*neg_b=*/false, b,
      /*c_mod=*/(short)0, c, /*reuse_a=*/false, /*reuse_b=*/false);
}

// ---------------- fragment-contiguous LDS swizzle ----------------
// A fragment (16x32): lane(h=lane>>4, r=lane&15) holds row r, elem i:
//   i<8  -> K = 8h+i ; i>=8 -> K = 16+8h+(i-8)       (ISA 7.12.2)
// B fragment (32x16): lane(h,n) holds col n, elem i: same K mapping as A
// Tiles stored so each lane's 16 elems are 32 contiguous bytes:
//   slot = (chunk*32 + lane)*16 + i  with chunk = (k>>5)*NT + tileIdx.
__device__ __forceinline__ int a_idx(int r, int k, int RT) {
  const int kk = k & 31;
  const int h = (kk >> 3) & 1;
  const int i = (kk & 7) | ((kk & 16) >> 1);
  return ((((k >> 5) * RT + (r >> 4)) * 32) + h * 16 + (r & 15)) * 16 + i;
}
__device__ __forceinline__ int b_idx(int k, int c, int CT) {
  const int kk = k & 31;
  const int h = (kk >> 3) & 1;
  const int i = (kk & 7) | ((kk & 16) >> 1);
  return ((((k >> 5) * CT + (c >> 4)) * 32) + h * 16 + (c & 15)) * 16 + i;
}
// fragment load: one aligned 32-byte LDS read (2x ds_load_b128)
__device__ __forceinline__ v16bf frag_ld(const unsigned short* base, int chunk,
                                         int lane) {
  return *reinterpret_cast<const v16bf*>(base + (chunk * 32 + lane) * 16);
}

// ---------------- RMSNorm ----------------
__global__ __launch_bounds__(256)
void rmsnorm_kernel(const float* __restrict__ x, const float* __restrict__ w,
                    float* __restrict__ out) {
  const int t = blockIdx.x;
  const float* xr = x + (long long)t * H;
  float ss = 0.f;
  for (int i = threadIdx.x; i < H; i += 256) { float v = xr[i]; ss += v * v; }
  __shared__ float red[256];
  red[threadIdx.x] = ss;
  __syncthreads();
  for (int s = 128; s > 0; s >>= 1) {
    if (threadIdx.x < s) red[threadIdx.x] += red[threadIdx.x + s];
    __syncthreads();
  }
  const float scale = rsqrtf(red[0] / (float)H + 1e-6f);
  for (int i = threadIdx.x; i < H; i += 256)
    out[(long long)t * H + i] = xr[i] * scale * w[i];
}

// ---------------- generic WMMA GEMM: C = A(MxK) @ B(KxN) [+ R] ----------------
// 128 threads (4 waves, 2x2 wave grid), block tile 64x64, K-step 32,
// ping-pong LDS (one barrier per step, staging overlaps WMMA).
// batched over blockIdx.z with strides sA/sB/sC.  M%64==0, N%64==0, K%32==0.
__global__ __launch_bounds__(128)
void gemm_kernel(const float* __restrict__ A, const float* __restrict__ B,
                 float* __restrict__ C, const float* __restrict__ R,
                 int M, int N, int Kd,
                 long long sA, long long sB, long long sC) {
  A += (long long)blockIdx.z * sA;
  B += (long long)blockIdx.z * sB;
  C += (long long)blockIdx.z * sC;
  __shared__ __align__(32) unsigned short As[2][2048];   // 64x32 swizzled (RT=4)
  __shared__ __align__(32) unsigned short Bs[2][2048];   // 32x64 swizzled (CT=4)
  const int tid = threadIdx.x, lane = tid & 31, wave = tid >> 5;
  const int wm = wave & 1, wn = wave >> 1;
  const int rowB = blockIdx.y * 64, colB = blockIdx.x * 64;

  auto stage = [&](int k0, int bufi) {
    // A tile: pairs along K (coalesced 8B/thread), packed ds_store_b32
    for (int i = tid; i < 64 * 16; i += 128) {
      int r = i >> 4, k2 = (i & 15) * 2;
      const float* src = &A[(rowB + r) * Kd + k0 + k2];   // in-batch: int math
      *(unsigned*)&As[bufi][a_idx(r, k2, 4)] = pack2bf(src[0], src[1]);
    }
    // B tile: pairs along K (two coalesced row reads), packed ds_store_b32
    for (int i = tid; i < 16 * 64; i += 128) {
      int k2 = (i >> 6) * 2, c = i & 63;
      float b0 = B[(k0 + k2) * N + colB + c];
      float b1 = B[(k0 + k2 + 1) * N + colB + c];
      *(unsigned*)&Bs[bufi][b_idx(k2, c, 4)] = pack2bf(b0, b1);
    }
  };

  v8f acc[2][2] = {};
  const int nk = Kd >> 5;
  stage(0, 0);
  __syncthreads();
  for (int t = 0; t < nk; ++t) {
    const int cur = t & 1;
    if (t + 1 < nk) stage((t + 1) * 32, cur ^ 1);   // overlap with WMMA below
    v16bf a0 = frag_ld(As[cur], wm * 2 + 0, lane);
    v16bf a1 = frag_ld(As[cur], wm * 2 + 1, lane);
    v16bf b0 = frag_ld(Bs[cur], wn * 2 + 0, lane);
    v16bf b1 = frag_ld(Bs[cur], wn * 2 + 1, lane);
    acc[0][0] = wmma_bf16(a0, b0, acc[0][0]);
    acc[0][1] = wmma_bf16(a0, b1, acc[0][1]);
    acc[1][0] = wmma_bf16(a1, b0, acc[1][0]);
    acc[1][1] = wmma_bf16(a1, b1, acc[1][1]);
    __syncthreads();
  }
  const int h = lane >> 4, n = lane & 15;
#pragma unroll
  for (int i = 0; i < 2; ++i)
#pragma unroll
    for (int j = 0; j < 2; ++j)
#pragma unroll
      for (int jv = 0; jv < 8; ++jv) {
        int row = rowB + wm * 32 + i * 16 + jv + 8 * h;
        int col = colB + wn * 32 + j * 16 + n;
        float val = acc[i][j][jv];
        if (R) val += R[row * N + col];
        C[row * N + col] = val;
      }
}

// ---------------- dual GEMM + SiLU-gate: C = silu(A@B1) * (A@B2) ----------------
__global__ __launch_bounds__(128)
void dualgemm_silu_kernel(const float* __restrict__ A,
                          const float* __restrict__ B1,
                          const float* __restrict__ B2,
                          float* __restrict__ C,
                          int M, int N, int Kd,
                          long long sA, long long sB, long long sC) {
  A  += (long long)blockIdx.z * sA;
  B1 += (long long)blockIdx.z * sB;
  B2 += (long long)blockIdx.z * sB;
  C  += (long long)blockIdx.z * sC;
  __shared__ __align__(32) unsigned short As[2][2048];
  __shared__ __align__(32) unsigned short Bs1[2][2048];
  __shared__ __align__(32) unsigned short Bs2[2][2048];
  const int tid = threadIdx.x, lane = tid & 31, wave = tid >> 5;
  const int wm = wave & 1, wn = wave >> 1;
  const int rowB = blockIdx.y * 64, colB = blockIdx.x * 64;

  auto stage = [&](int k0, int bufi) {
    for (int i = tid; i < 64 * 16; i += 128) {
      int r = i >> 4, k2 = (i & 15) * 2;
      const float* src = &A[(rowB + r) * Kd + k0 + k2];
      *(unsigned*)&As[bufi][a_idx(r, k2, 4)] = pack2bf(src[0], src[1]);
    }
    for (int i = tid; i < 16 * 64; i += 128) {
      int k2 = (i >> 6) * 2, c = i & 63;
      int g0 = (k0 + k2) * N + colB + c;
      int g1 = g0 + N;
      int bi = b_idx(k2, c, 4);
      *(unsigned*)&Bs1[bufi][bi] = pack2bf(B1[g0], B1[g1]);
      *(unsigned*)&Bs2[bufi][bi] = pack2bf(B2[g0], B2[g1]);
    }
  };

  v8f acc1[2][2] = {};
  v8f acc2[2][2] = {};
  const int nk = Kd >> 5;
  stage(0, 0);
  __syncthreads();
  for (int t = 0; t < nk; ++t) {
    const int cur = t & 1;
    if (t + 1 < nk) stage((t + 1) * 32, cur ^ 1);
    v16bf a0 = frag_ld(As[cur], wm * 2 + 0, lane);
    v16bf a1 = frag_ld(As[cur], wm * 2 + 1, lane);
#pragma unroll
    for (int j = 0; j < 2; ++j) {
      v16bf b1 = frag_ld(Bs1[cur], wn * 2 + j, lane);
      v16bf b2 = frag_ld(Bs2[cur], wn * 2 + j, lane);
      acc1[0][j] = wmma_bf16(a0, b1, acc1[0][j]);
      acc1[1][j] = wmma_bf16(a1, b1, acc1[1][j]);
      acc2[0][j] = wmma_bf16(a0, b2, acc2[0][j]);
      acc2[1][j] = wmma_bf16(a1, b2, acc2[1][j]);
    }
    __syncthreads();
  }
  const int h = lane >> 4, n = lane & 15;
#pragma unroll
  for (int i = 0; i < 2; ++i)
#pragma unroll
    for (int j = 0; j < 2; ++j)
#pragma unroll
      for (int jv = 0; jv < 8; ++jv) {
        int row = rowB + wm * 32 + i * 16 + jv + 8 * h;
        int col = colB + wn * 32 + j * 16 + n;
        C[row * N + col] = silu(acc1[i][j][jv]) * acc2[i][j][jv];
      }
}

// ---------------- RoPE (applied in-place to q and k) ----------------
__global__ void rope_kernel(float* __restrict__ q, float* __restrict__ k) {
  const int s = blockIdx.x, head = blockIdx.y, d = threadIdx.x;  // d in [0,32)
  const float inv = __powf(10000.0f, -((float)(2 * d) / (float)HD));
  const float fr = (float)s * inv;
  const float c = __cosf(fr), sn = __sinf(fr);
  float* base = (head < NH) ? (q + s * (NH * HD) + head * HD)
                            : (k + s * (KVH * HD) + (head - NH) * HD);
  const float x0 = base[d], x1 = base[d + 32];
  base[d]      = x0 * c - x1 * sn;
  base[d + 32] = x1 * c + x0 * sn;
}

// ---------------- flash attention: one (head, 64-q-row) block ----------------
// Qs/Ps: A-layout RT=4, KC=2; Ks/Vs: B-layout CT=4, KC=2 (all 4096 ushorts)
__global__ __launch_bounds__(128)
void attn_kernel(const float* __restrict__ q, const float* __restrict__ kk,
                 const float* __restrict__ vv, float* __restrict__ o) {
  const int qb = blockIdx.x, head = blockIdx.y;
  const int kvh = head / REP;
  const int tid = threadIdx.x, lane = tid & 31, wave = tid >> 5;
  const int h = lane >> 4, n = lane & 15;
  __shared__ __align__(32) unsigned short Qs[4096], Ks[4096], Vs[4096], Ps[4096];
  __shared__ float Ss[64 * 65];
  __shared__ float mrow[64], lrow[64], frow[64];

  for (int i = tid; i < 64 * 32; i += 128) {   // Q tile, pairs along HD
    int r = i >> 5, d2 = (i & 31) * 2;
    const float* src = &q[(qb * 64 + r) * (NH * HD) + head * HD + d2];
    *(unsigned*)&Qs[a_idx(r, d2, 4)] = pack2bf(src[0], src[1]);
  }
  if (tid < 64) { mrow[tid] = -1e30f; lrow[tid] = 0.f; }
  v8f oacc[4] = {};
  __syncthreads();

  for (int kb = 0; kb <= qb; ++kb) {
    for (int i = tid; i < 64 * 32; i += 128) {
      int t = i >> 5, d2 = (i & 31) * 2;
      int gi = (kb * 64 + t) * (KVH * HD) + kvh * HD + d2;
      // K^T tile: B-layout, contraction dim = HD -> pairs pack along HD
      *(unsigned*)&Ks[b_idx(d2, t, 4)] = pack2bf(kk[gi], kk[gi + 1]);
      // V tile: B-layout, contraction dim = token -> scalar stores
      Vs[b_idx(t, d2, 4)]     = f2bfu(vv[gi]);
      Vs[b_idx(t, d2 + 1, 4)] = f2bfu(vv[gi + 1]);
    }
    __syncthreads();

    v8f sacc[4] = {};
#pragma unroll
    for (int kc = 0; kc < 2; ++kc) {
      v16bf a = frag_ld(Qs, kc * 4 + wave, lane);
#pragma unroll
      for (int j = 0; j < 4; ++j) {
        v16bf b = frag_ld(Ks, kc * 4 + j, lane);
        sacc[j] = wmma_bf16(a, b, sacc[j]);
      }
    }
#pragma unroll
    for (int j = 0; j < 4; ++j)
#pragma unroll
      for (int jv = 0; jv < 8; ++jv) {
        int row = wave * 16 + jv + 8 * h, col = j * 16 + n;
        float val = sacc[j][jv] * 0.125f;               // 1/sqrt(HD)
        if (kb == qb && col > row) val += -1.0e9f;      // causal mask
        Ss[row * 65 + col] = val;
      }
    __syncthreads();

    if (tid < 64) {  // online softmax, one row per thread
      float mold = mrow[tid], mloc = -1e30f;
      for (int c = 0; c < 64; ++c) mloc = fmaxf(mloc, Ss[tid * 65 + c]);
      float mnew = fmaxf(mold, mloc);
      float f = __expf(mold - mnew);
      float sum = 0.f;
      for (int c = 0; c < 64; c += 2) {
        float p0 = __expf(Ss[tid * 65 + c] - mnew);
        float p1 = __expf(Ss[tid * 65 + c + 1] - mnew);
        sum += p0 + p1;
        *(unsigned*)&Ps[a_idx(tid, c, 4)] = pack2bf(p0, p1);
      }
      lrow[tid] = lrow[tid] * f + sum;
      mrow[tid] = mnew;
      frow[tid] = f;
    }
    __syncthreads();

#pragma unroll
    for (int j = 0; j < 4; ++j)
#pragma unroll
      for (int jv = 0; jv < 8; ++jv)
        oacc[j][jv] *= frow[wave * 16 + jv + 8 * h];
#pragma unroll
    for (int kc = 0; kc < 2; ++kc) {
      v16bf a = frag_ld(Ps, kc * 4 + wave, lane);
#pragma unroll
      for (int j = 0; j < 4; ++j) {
        v16bf b = frag_ld(Vs, kc * 4 + j, lane);
        oacc[j] = wmma_bf16(a, b, oacc[j]);
      }
    }
    __syncthreads();
  }

#pragma unroll
  for (int j = 0; j < 4; ++j)
#pragma unroll
    for (int jv = 0; jv < 8; ++jv) {
      int row = wave * 16 + jv + 8 * h, col = j * 16 + n;
      o[(qb * 64 + row) * (NH * HD) + head * HD + col] = oacc[j][jv] / lrow[row];
    }
}

// ---------------- top-k + softmax (+ per-token expert bitmask) ----------------
__global__ __launch_bounds__(256)
void topk_kernel(const float* __restrict__ logits, int* __restrict__ topi,
                 float* __restrict__ topp, unsigned long long* __restrict__ emask) {
  const int t = blockIdx.x * blockDim.x + threadIdx.x;
  if (t >= T) return;
  float lv[E];
#pragma unroll
  for (int e = 0; e < E; ++e) lv[e] = logits[t * E + e];
  float vals[TOPK]; int ids[TOPK];
  unsigned long long mask = 0ull;
  for (int k2 = 0; k2 < TOPK; ++k2) {
    float best = -1e30f; int bi = 0;
#pragma unroll
    for (int e = 0; e < E; ++e)
      if (!((mask >> e) & 1ull) && lv[e] > best) { best = lv[e]; bi = e; }
    vals[k2] = best; ids[k2] = bi; mask |= 1ull << bi;
  }
  const float mx = vals[0];
  float s = 0.f, p[TOPK];
#pragma unroll
  for (int k2 = 0; k2 < TOPK; ++k2) { p[k2] = __expf(vals[k2] - mx); s += p[k2]; }
  const float inv = 1.0f / s;
#pragma unroll
  for (int k2 = 0; k2 < TOPK; ++k2) {
    topi[t * TOPK + k2] = ids[k2];
    topp[t * TOPK + k2] = p[k2] * inv;
  }
  emask[t] = mask;
}

// ---------------- parallel capacity routing ----------------
// Within a token the top-k experts are distinct, so the flat-order rank of
// entry (t,k) for expert e = #{t' < t : token t' chose e}.  One 1024-thread
// block: thread (e, seg) counts expert e over its 64-token segment, LDS
// prefix over segments, then writes ranks.  Exactly matches the reference's
// ordered cumsum drop rule.
__global__ __launch_bounds__(1024)
void route_kernel(const unsigned long long* __restrict__ emask,
                  int* __restrict__ posTE) {
  const int e = threadIdx.x & 63, seg = threadIdx.x >> 6;   // 16 segments
  __shared__ int segcnt[16][64];
  int cnt = 0;
  for (int t = seg * 64; t < seg * 64 + 64; ++t)
    cnt += (int)((emask[t] >> e) & 1ull);
  segcnt[seg][e] = cnt;
  __syncthreads();
  int run = 0;
  for (int s2 = 0; s2 < seg; ++s2) run += segcnt[s2][e];
  for (int t = seg * 64; t < seg * 64 + 64; ++t)
    if ((emask[t] >> e) & 1ull) { posTE[t * 64 + e] = run; ++run; }
}

__global__ __launch_bounds__(256)
void finalize_route_kernel(const int* __restrict__ topi,
                           const float* __restrict__ topp,
                           const int* __restrict__ posTE,
                           int* __restrict__ pos, float* __restrict__ coef) {
  const int i = blockIdx.x * 256 + threadIdx.x;
  if (i >= T * TOPK) return;
  const int t = i >> 3;
  const int p = posTE[t * 64 + topi[i]];
  pos[i] = p;
  coef[i] = (p < CAP) ? topp[i] : 0.f;
}

// ---------------- scatter tokens into expert capacity buffer ----------------
__global__ __launch_bounds__(256)
void scatter_kernel(const float* __restrict__ xn2, const int* __restrict__ topi,
                    const int* __restrict__ pos, float* __restrict__ buf) {
  const int idx = blockIdx.x;          // (t,k) pair
  const int p = pos[idx];
  if (p >= CAP) return;
  const int e = topi[idx], t = idx >> 3;
  float* dst = buf + (e * CAP + p) * H;
  const float* src = xn2 + t * H;
  for (int i = threadIdx.x; i < H; i += 256) dst[i] = src[i];
}

// ---------------- final combine: out = h + shared + sum coef*y ----------------
__global__ __launch_bounds__(256)
void combine_kernel(const float* __restrict__ hres, const float* __restrict__ ysh,
                    const int* __restrict__ topi, const int* __restrict__ pos,
                    const float* __restrict__ coef, const float* __restrict__ ybuf,
                    float* __restrict__ out) {
  const int t = blockIdx.x;
  for (int i = threadIdx.x; i < H; i += 256) {
    float acc = hres[t * H + i] + ysh[t * H + i];
#pragma unroll
    for (int k2 = 0; k2 < TOPK; ++k2) {
      const int idx = t * TOPK + k2;
      const float c = coef[idx];
      if (c != 0.f) acc += c * ybuf[(topi[idx] * CAP + pos[idx]) * H + i];
    }
    out[t * H + i] = acc;
  }
}

// ---------------- host launcher ----------------
extern "C" void kernel_launch(void* const* d_in, const int* in_sizes, int n_in,
                              void* d_out, int out_size, void* d_ws, size_t ws_size,
                              hipStream_t stream) {
  (void)in_sizes; (void)n_in; (void)out_size; (void)ws_size;
  const float* hidden = (const float*)d_in[0];
  // d_in[1] = attention_mask: pure causal, applied analytically in attn_kernel
  const float* w_ln1 = (const float*)d_in[2];
  const float* w_ln2 = (const float*)d_in[3];
  const float* Wq    = (const float*)d_in[4];
  const float* Wk    = (const float*)d_in[5];
  const float* Wv    = (const float*)d_in[6];
  const float* Wo    = (const float*)d_in[7];
  const float* Wgate = (const float*)d_in[8];
  const float* Weg   = (const float*)d_in[9];
  const float* Weu   = (const float*)d_in[10];
  const float* Wed   = (const float*)d_in[11];
  const float* Wsg   = (const float*)d_in[12];
  const float* Wsu   = (const float*)d_in[13];
  const float* Wsd   = (const float*)d_in[14];
  float* out = (float*)d_out;

  char* w = (char*)d_ws;
  auto alloc = [&](size_t bytes) -> void* {
    void* p = (void*)w;
    w += (bytes + 255) & ~(size_t)255;
    return p;
  };
  float* xn1    = (float*)alloc((size_t)T * H * 4);
  float* qb     = (float*)alloc((size_t)T * H * 4);
  float* kb     = (float*)alloc((size_t)T * KVH * HD * 4);
  float* vb     = (float*)alloc((size_t)T * KVH * HD * 4);
  float* ob     = (float*)alloc((size_t)T * H * 4);
  float* hb     = (float*)alloc((size_t)T * H * 4);
  float* xn2    = (float*)alloc((size_t)T * H * 4);
  float* logits = (float*)alloc((size_t)T * E * 4);
  int*   topi   = (int*)alloc((size_t)T * TOPK * 4);
  float* topp   = (float*)alloc((size_t)T * TOPK * 4);
  unsigned long long* emask = (unsigned long long*)alloc((size_t)T * 8);
  int*   posTE  = (int*)alloc((size_t)T * E * 4);
  int*   pos    = (int*)alloc((size_t)T * TOPK * 4);
  float* coef   = (float*)alloc((size_t)T * TOPK * 4);
  float* buf    = (float*)alloc((size_t)E * CAP * H * 4);
  float* act    = (float*)alloc((size_t)E * CAP * I * 4);
  float* ybuf   = (float*)alloc((size_t)E * CAP * H * 4);
  float* acts   = (float*)alloc((size_t)T * IS * 4);
  float* ysh    = (float*)alloc((size_t)T * H * 4);

  // ---- attention path ----
  rmsnorm_kernel<<<T, 256, 0, stream>>>(hidden, w_ln1, xn1);
  gemm_kernel<<<dim3(H / 64, T / 64, 1), 128, 0, stream>>>(xn1, Wq, qb, nullptr, T, H, H, 0, 0, 0);
  gemm_kernel<<<dim3((KVH * HD) / 64, T / 64, 1), 128, 0, stream>>>(xn1, Wk, kb, nullptr, T, KVH * HD, H, 0, 0, 0);
  gemm_kernel<<<dim3((KVH * HD) / 64, T / 64, 1), 128, 0, stream>>>(xn1, Wv, vb, nullptr, T, KVH * HD, H, 0, 0, 0);
  rope_kernel<<<dim3(S, NH + KVH), 32, 0, stream>>>(qb, kb);
  attn_kernel<<<dim3(S / 64, NH), 128, 0, stream>>>(qb, kb, vb, ob);
  gemm_kernel<<<dim3(H / 64, T / 64, 1), 128, 0, stream>>>(ob, Wo, hb, hidden, T, H, H, 0, 0, 0);

  // ---- MoE path ----
  rmsnorm_kernel<<<T, 256, 0, stream>>>(hb, w_ln2, xn2);
  gemm_kernel<<<dim3(1, T / 64, 1), 128, 0, stream>>>(xn2, Wgate, logits, nullptr, T, E, H, 0, 0, 0);
  topk_kernel<<<(T + 255) / 256, 256, 0, stream>>>(logits, topi, topp, emask);
  route_kernel<<<1, 1024, 0, stream>>>(emask, posTE);
  finalize_route_kernel<<<(T * TOPK + 255) / 256, 256, 0, stream>>>(topi, topp, posTE, pos, coef);
  (void)hipMemsetAsync(buf, 0, (size_t)E * CAP * H * 4, stream);
  scatter_kernel<<<T * TOPK, 256, 0, stream>>>(xn2, topi, pos, buf);

  // routed experts: act = silu(buf@Weg)*(buf@Weu), y = act@Wed  (batched over E)
  dualgemm_silu_kernel<<<dim3(I / 64, CAP / 64, E), 128, 0, stream>>>(
      buf, Weg, Weu, act, CAP, I, H,
      (long long)CAP * H, (long long)H * I, (long long)CAP * I);
  gemm_kernel<<<dim3(H / 64, CAP / 64, E), 128, 0, stream>>>(
      act, Wed, ybuf, nullptr, CAP, H, I,
      (long long)CAP * I, (long long)I * H, (long long)CAP * H);

  // shared expert
  dualgemm_silu_kernel<<<dim3(IS / 64, T / 64, 1), 128, 0, stream>>>(
      xn2, Wsg, Wsu, acts, T, IS, H, 0, 0, 0);
  gemm_kernel<<<dim3(H / 64, T / 64, 1), 128, 0, stream>>>(
      acts, Wsd, ysh, nullptr, T, H, IS, 0, 0, 0);

  // final combine
  combine_kernel<<<T, 256, 0, stream>>>(hb, ysh, topi, pos, coef, ybuf, out);
}
